// PathAgents_64682207478427
// MI455X (gfx1250) — compile-verified, matrix operand order checked by reading, not run on hardware
//
#include <hip/hip_runtime.h>
#include <hip/hip_bf16.h>

// ---------------------------------------------------------------------------
// PathAgents forward for MI455X (gfx1250, wave32, WMMA + TDM).
// Heavy GEMMs: v_wmma_f32_16x16x32_f16, tiles DMA'd to LDS by the Tensor Data
// Mover (tensor_load_to_lds, TENSORcnt, double-buffered) when available.
// ---------------------------------------------------------------------------

typedef __attribute__((ext_vector_type(16))) _Float16 v16h;
typedef __attribute__((ext_vector_type(8)))  _Float16 v8h;
typedef __attribute__((ext_vector_type(8)))  float    v8f;
typedef __attribute__((ext_vector_type(4)))  unsigned int u32x4;
typedef __attribute__((ext_vector_type(8)))  int      i32x8;
typedef __attribute__((ext_vector_type(4)))  int      i32x4;

#define V_  8
#define T_  2048      // B*L
#define L_  1024
#define B_  2
#define DM_ 512
#define DI_ 1024
#define R0_ 16384     // B*L*V rows for embedding

#if defined(__gfx1250__) && __has_builtin(__builtin_amdgcn_tensor_load_to_lds) && \
    __has_builtin(__builtin_amdgcn_s_wait_tensorcnt)
#define HAVE_TDM 1
#else
#define HAVE_TDM 0
#endif

#if HAVE_TDM
// Issue a 2D TDM load: rows x 32 f16 elements, global row stride = K elements,
// LDS rows padded to 48 halves (pad 8 dwords after every 16 dwords stored).
__device__ __forceinline__ void tdm_tile_load(const _Float16* gptr, unsigned lds_byte,
                                              int rows, int K)
{
  const unsigned long long ga = (unsigned long long)gptr;
  u32x4 g0;
  g0[0] = 1u;                                    // count=1, user descriptor
  g0[1] = lds_byte;                              // lds_addr
  g0[2] = (unsigned)ga;                          // global_addr[31:0]
  g0[3] = (unsigned)((ga >> 32) & 0x01FFFFFFu)   // global_addr[56:32]
        | (2u << 30);                            // type = 2 (image)
  i32x8 g1;
  g1[0] = (int)((1u << 16)                       // data_size = 2 bytes
              | (1u << 20)                       // pad_enable
              | (3u << 22)                       // pad_interval: 16 DWORDs
              | (7u << 25));                     // pad_amount: 8 DWORDs
  g1[1] = (int)(32u << 16);                      // tensor_dim0 = 32
  g1[2] = (int)((unsigned)rows << 16);           // tensor_dim1[15:0] = rows
  g1[3] = (int)(32u << 16);                      // tile_dim0 = 32
  g1[4] = rows;                                  // tile_dim1 = rows (tile_dim2=0)
  g1[5] = K;                                     // tensor_dim0_stride[31:0]
  g1[6] = 0;
  g1[7] = 0;
  const i32x4 z4 = {0, 0, 0, 0};
#if __clang_major__ >= 23
  const i32x8 z8 = {0, 0, 0, 0, 0, 0, 0, 0};
  __builtin_amdgcn_tensor_load_to_lds(g0, g1, z4, z4, z8, 0);
#else
  __builtin_amdgcn_tensor_load_to_lds(g0, g1, z4, z4, 0);
#endif
}
#endif

// ---------------------------------------------------------------------------
// Generic WMMA GEMM: C[M,N] = A[M,K] @ W[N,K]^T (+ bias), batched over z.
// A, W row-major f16; C f32. Requires M%64==0, N%128==0, K%32==0.
// Block 256 threads = 8 waves (4 along M x 2 along N), tile 64x128.
// ---------------------------------------------------------------------------
__global__ __launch_bounds__(256)
void gemm_f16f32(const _Float16* __restrict__ A, const _Float16* __restrict__ W,
                 float* __restrict__ C, const float* __restrict__ bias,
                 int M, int N, int K,
                 long long sA, long long sW, long long sC)
{
  const int bN = blockIdx.x * 128;
  const int bM = blockIdx.y * 64;
  A += (long long)blockIdx.z * sA;
  W += (long long)blockIdx.z * sW;
  C += (long long)blockIdx.z * sC;

  const int tid  = threadIdx.x;
  const int lane = tid & 31;
  const int wv   = tid >> 5;   // 0..7
  const int wm   = wv & 3;     // 4 waves along M
  const int wn   = wv >> 2;    // 2 waves along N

  const int fm = lane & 15;
  const int kb = (lane >> 4) << 3;   // 0 or 8

  v8f acc[4] = {};

#if HAVE_TDM
  // double-buffered LDS, tiles DMA'd by the Tensor Data Mover
  __shared__ __align__(16) _Float16 As[2][64 * 48];
  __shared__ __align__(16) _Float16 Bs[2][128 * 48];
  const unsigned ldsA0 = (unsigned)(unsigned long long)(void*)&As[0][0];
  const unsigned ldsA1 = (unsigned)(unsigned long long)(void*)&As[1][0];
  const unsigned ldsB0 = (unsigned)(unsigned long long)(void*)&Bs[0][0];
  const unsigned ldsB1 = (unsigned)(unsigned long long)(void*)&Bs[1][0];
  const bool issuer = (wv == 0);     // wave-uniform

  const int nk = K >> 5;
  if (issuer) {
    tdm_tile_load(A + (long long)bM * K, ldsA0, 64, K);
    tdm_tile_load(W + (long long)bN * K, ldsB0, 128, K);
  }
  for (int ik = 0; ik < nk; ++ik) {
    if (issuer) __builtin_amdgcn_s_wait_tensorcnt(0);
    __syncthreads();
    if (issuer && (ik + 1) < nk) {      // prefetch next tile into other buffer
      const long long k0n = (long long)(ik + 1) << 5;
      tdm_tile_load(A + (long long)bM * K + k0n, ((ik + 1) & 1) ? ldsA1 : ldsA0, 64, K);
      tdm_tile_load(W + (long long)bN * K + k0n, ((ik + 1) & 1) ? ldsB1 : ldsB0, 128, K);
    }
    const _Float16* Ab = As[ik & 1];
    const _Float16* Bb = Bs[ik & 1];
    v16h af;
    {
      const _Float16* p = Ab + (wm * 16 + fm) * 48 + kb;
      const v8h lo = *(const v8h*)(p);
      const v8h hi = *(const v8h*)(p + 16);
      #pragma unroll
      for (int i = 0; i < 8; ++i) { af[i] = lo[i]; af[8 + i] = hi[i]; }
    }
    v16h bf[4];
    #pragma unroll
    for (int t = 0; t < 4; ++t) {
      const _Float16* p = Bb + (wn * 64 + t * 16 + fm) * 48 + kb;
      const v8h lo = *(const v8h*)(p);
      const v8h hi = *(const v8h*)(p + 16);
      #pragma unroll
      for (int i = 0; i < 8; ++i) { bf[t][i] = lo[i]; bf[t][8 + i] = hi[i]; }
    }
    #pragma unroll
    for (int t = 0; t < 4; ++t)
      acc[t] = __builtin_amdgcn_wmma_f32_16x16x32_f16(
          false, af, false, bf[t], (short)0, acc[t], false, false);
    __syncthreads();
  }
#else
  // fallback: register-staged tiles (grouped loads -> single loadcnt wait)
  __shared__ __align__(16) _Float16 As[64 * 48];
  __shared__ __align__(16) _Float16 Bs[128 * 48];
  const int arow = tid >> 2, aseg = tid & 3;
  for (int k0 = 0; k0 < K; k0 += 32) {
    const uint4 av  = *(const uint4*)(A + (long long)(bM + arow) * K + k0 + aseg * 8);
    const uint4 bv0 = *(const uint4*)(W + (long long)(bN + arow) * K + k0 + aseg * 8);
    const uint4 bv1 = *(const uint4*)(W + (long long)(bN + 64 + arow) * K + k0 + aseg * 8);
    *(uint4*)(As + arow * 48 + aseg * 8)        = av;
    *(uint4*)(Bs + arow * 48 + aseg * 8)        = bv0;
    *(uint4*)(Bs + (64 + arow) * 48 + aseg * 8) = bv1;
    __syncthreads();
    v16h af;
    {
      const _Float16* p = As + (wm * 16 + fm) * 48 + kb;
      const v8h lo = *(const v8h*)(p);
      const v8h hi = *(const v8h*)(p + 16);
      #pragma unroll
      for (int i = 0; i < 8; ++i) { af[i] = lo[i]; af[8 + i] = hi[i]; }
    }
    v16h bf[4];
    #pragma unroll
    for (int t = 0; t < 4; ++t) {
      const _Float16* p = Bs + (wn * 64 + t * 16 + fm) * 48 + kb;
      const v8h lo = *(const v8h*)(p);
      const v8h hi = *(const v8h*)(p + 16);
      #pragma unroll
      for (int i = 0; i < 8; ++i) { bf[t][i] = lo[i]; bf[t][8 + i] = hi[i]; }
    }
    #pragma unroll
    for (int t = 0; t < 4; ++t)
      acc[t] = __builtin_amdgcn_wmma_f32_16x16x32_f16(
          false, af, false, bf[t], (short)0, acc[t], false, false);
    __syncthreads();
  }
#endif

  // writeback: lane holds n = lane&15, rows m = (lane>>4)*8 + r
  const int n0   = bN + wn * 64 + (lane & 15);
  const int mrow = bM + wm * 16 + ((lane >> 4) << 3);
  #pragma unroll
  for (int t = 0; t < 4; ++t) {
    const int n = n0 + t * 16;
    const float bv = bias ? bias[n] : 0.f;
    #pragma unroll
    for (int r = 0; r < 8; ++r)
      C[(long long)(mrow + r) * N + n] = acc[t][r] + bv;
  }
}

// ---------------------------------------------------------------------------
__global__ void f32_to_f16(const float* __restrict__ in, _Float16* __restrict__ out, int n)
{
  const int i = blockIdx.x * 256 + threadIdx.x;
  if (i < n) out[i] = (_Float16)in[i];
}

// x * bn_g / sqrt(1+1e-5) + bn_b  ->  f16
__global__ void prep_x(const float* __restrict__ x, const float* __restrict__ g,
                       const float* __restrict__ b, _Float16* __restrict__ out)
{
  const int i = blockIdx.x * 256 + threadIdx.x;   // R0*1024
  const int c = i & 1023;
  out[i] = (_Float16)(x[i] * (g[c] * 0.99999500003749981f) + b[c]);
}

// ---------------------------------------------------------------------------
__device__ __forceinline__ float block_sum256(float v, float* red)
{
  __syncthreads();
  #pragma unroll
  for (int off = 16; off; off >>= 1) v += __shfl_xor(v, off, 32);
  if ((threadIdx.x & 31) == 0) red[threadIdx.x >> 5] = v;
  __syncthreads();
  return red[0] + red[1] + red[2] + red[3] + red[4] + red[5] + red[6] + red[7];
}

// xe row + emb_b -> LN(1e-6) -> GLU -> h in (V,T,512) layout
__global__ __launch_bounds__(256)
void embed_post(const float* __restrict__ xe, const float* __restrict__ bias,
                const float* __restrict__ g, const float* __restrict__ b,
                float* __restrict__ h)
{
  __shared__ float row[1024];
  __shared__ float red[8];
  const int r = blockIdx.x, tid = threadIdx.x;
  float part = 0.f;
  for (int c = tid; c < 1024; c += 256) {
    const float val = xe[(size_t)r * 1024 + c] + bias[c];
    row[c] = val; part += val;
  }
  const float mu = block_sum256(part, red) * (1.f / 1024.f);
  float vp = 0.f;
  for (int c = tid; c < 1024; c += 256) { const float d0 = row[c] - mu; vp += d0 * d0; }
  const float var = block_sum256(vp, red) * (1.f / 1024.f);
  const float inv = rsqrtf(var + 1e-6f);
  const int v = r & 7, t = r >> 3;
  for (int d0 = tid; d0 < 512; d0 += 256) {
    const float a  = (row[d0]       - mu) * inv * g[d0]       + b[d0];
    const float gt = (row[d0 + 512] - mu) * inv * g[d0 + 512] + b[d0 + 512];
    h[((size_t)v * T_ + t) * 512 + d0] = a * (gt / (1.f + __expf(-gt)));
  }
}

// per-view LN (eps 1e-5) of h(V,T,512) -> f16
__global__ __launch_bounds__(256)
void ln512_f16(const float* __restrict__ hin, const float* __restrict__ g,
               const float* __restrict__ b, _Float16* __restrict__ out)
{
  __shared__ float row[512];
  __shared__ float red[8];
  const int r = blockIdx.x, tid = threadIdx.x;
  const int v = r >> 11;    // T = 2048
  float part = 0.f;
  for (int c = tid; c < 512; c += 256) {
    const float val = hin[(size_t)r * 512 + c];
    row[c] = val; part += val;
  }
  const float mu = block_sum256(part, red) * (1.f / 512.f);
  float vp = 0.f;
  for (int c = tid; c < 512; c += 256) { const float d0 = row[c] - mu; vp += d0 * d0; }
  const float var = block_sum256(vp, red) * (1.f / 512.f);
  const float inv = rsqrtf(var + 1e-5f);
  for (int c = tid; c < 512; c += 256)
    out[(size_t)r * 512 + c] =
        (_Float16)((row[c] - mu) * inv * g[v * 512 + c] + b[v * 512 + c]);
}

// depthwise causal conv(4) + bias + SiLU over xp = xz[..., :1024]
__global__ void conv_silu(const float* __restrict__ xz, const float* __restrict__ cw,
                          const float* __restrict__ cb, float* __restrict__ xc)
{
  const long long i = (long long)blockIdx.x * 256 + threadIdx.x;  // V*T*1024
  const int d = (int)(i & 1023);
  const long long vt = i >> 10;
  const int t = (int)(vt & 2047);
  const int v = (int)(vt >> 11);
  const int l = t & 1023;
  const int bb = (t >> 10) & 1;
  float acc = cb[v * DI_ + d];
  const float* w = cw + ((size_t)v * DI_ + d) * 4;
  #pragma unroll
  for (int j = 0; j < 4; ++j) {
    const int l2 = l - 3 + j;
    if (l2 >= 0)
      acc += w[j] * xz[(((size_t)v * T_) + (size_t)bb * L_ + l2) * 2048 + d];
  }
  xc[i] = acc / (1.f + __expf(-acc));
}

// dbc[v,t,0:40] = xc_row @ xproj_w[v].T  (row staged in LDS, wave-per-output)
__global__ __launch_bounds__(256)
void xproj_k(const float* __restrict__ xc, const float* __restrict__ w,
             float* __restrict__ dbc)
{
  __shared__ float s[1024];
  const int r = blockIdx.x;           // v*T + t
  const int v = r >> 11;
  for (int c = threadIdx.x; c < 1024; c += 256) s[c] = xc[(size_t)r * 1024 + c];
  __syncthreads();
  const int lane = threadIdx.x & 31, wv = threadIdx.x >> 5;
  for (int o = wv; o < 40; o += 8) {
    const float* wr = w + ((size_t)v * 40 + o) * 1024;
    float p = 0.f;
    for (int d = lane; d < 1024; d += 32) p += s[d] * wr[d];
    #pragma unroll
    for (int off = 16; off; off >>= 1) p += __shfl_xor(p, off, 32);
    if (lane == 0) dbc[(size_t)r * 40 + o] = p;
  }
}

// delta = softplus(dbc[:,:32] @ dt_w[v].T + dt_b[v])
__global__ void delta_k(const float* __restrict__ dbc, const float* __restrict__ dtw,
                        const float* __restrict__ dtb, float* __restrict__ delta)
{
  const long long i = (long long)blockIdx.x * 256 + threadIdx.x;   // V*T*1024
  const int d = (int)(i & 1023);
  const long long vt = i >> 10;
  const int v = (int)(vt >> 11);
  const float* r = dbc + vt * 40;
  const float* w = dtw + ((size_t)v * DI_ + d) * 32;
  float acc = dtb[v * DI_ + d];
  #pragma unroll
  for (int j = 0; j < 32; ++j) acc += r[j] * w[j];
  delta[i] = (acc > 20.f) ? acc : log1pf(__expf(acc));
}

// selective scan: thread per (v,b,d); fused +D*xc and *silu(z); writes f16 for out-GEMM
__global__ __launch_bounds__(256)
void scan_k(const float* __restrict__ delta, const float* __restrict__ xc,
            const float* __restrict__ dbc, const float* __restrict__ xz,
            const float* __restrict__ Alog, const float* __restrict__ Dp,
            _Float16* __restrict__ yf16)
{
  const int i = blockIdx.x * 256 + threadIdx.x;  // V*B*1024 = 16384
  const int d  = i & 1023;
  const int vb = i >> 10;
  const int b  = vb & 1;
  const int v  = vb >> 1;
  float Am[4];
  #pragma unroll
  for (int s = 0; s < 4; ++s) Am[s] = -__expf(Alog[((size_t)v * DI_ + d) * 4 + s]);
  const float Dv = Dp[v * DI_ + d];
  float h0 = 0.f, h1 = 0.f, h2 = 0.f, h3 = 0.f;
  const size_t base = (size_t)v * T_ + (size_t)b * L_;
  for (int l = 0; l < L_; ++l) {
    const size_t row = base + l;
    const float dlt = delta[row * 1024 + d];
    const float xcv = xc[row * 1024 + d];
    const float* bc = dbc + row * 40;
    h0 = __expf(dlt * Am[0]) * h0 + dlt * bc[32] * xcv;
    h1 = __expf(dlt * Am[1]) * h1 + dlt * bc[33] * xcv;
    h2 = __expf(dlt * Am[2]) * h2 + dlt * bc[34] * xcv;
    h3 = __expf(dlt * Am[3]) * h3 + dlt * bc[35] * xcv;
    const float y = h0 * bc[36] + h1 * bc[37] + h2 * bc[38] + h3 * bc[39];
    const float zv = xz[row * 2048 + 1024 + d];
    const float yy = (y + Dv * xcv) * (zv / (1.f + __expf(-zv)));
    yf16[row * 1024 + d] = (_Float16)yy;
  }
}

// cross-view attention: block per (token, head), V=8, head_dim=64
__global__ __launch_bounds__(64)
void attn_k(const float* __restrict__ qkv, _Float16* __restrict__ o)
{
  __shared__ float q[8][64], k[8][64], vv[8][64], sc[8][8];
  const int t = blockIdx.x, hh = blockIdx.y, tid = threadIdx.x;
  for (int idx = tid; idx < 512; idx += 64) {
    const int vi = idx >> 6, c = idx & 63;
    const size_t row = ((size_t)vi * T_ + t) * 1536;
    q[vi][c]  = qkv[row + hh * 64 + c];
    k[vi][c]  = qkv[row + 512 + hh * 64 + c];
    vv[vi][c] = qkv[row + 1024 + hh * 64 + c];
  }
  __syncthreads();
  {
    const int i = tid >> 3, j = tid & 7;
    float p = 0.f;
    for (int c = 0; c < 64; ++c) p += q[i][c] * k[j][c];
    sc[i][j] = p * 0.125f;   // 1/sqrt(64)
  }
  __syncthreads();
  if (tid < 8) {
    float m = sc[tid][0];
    for (int j = 1; j < 8; ++j) m = fmaxf(m, sc[tid][j]);
    float s = 0.f;
    for (int j = 0; j < 8; ++j) { const float e = __expf(sc[tid][j] - m); sc[tid][j] = e; s += e; }
    const float inv = 1.f / s;
    for (int j = 0; j < 8; ++j) sc[tid][j] *= inv;
  }
  __syncthreads();
  for (int i = 0; i < 8; ++i) {
    float p = 0.f;
    #pragma unroll
    for (int j = 0; j < 8; ++j) p += sc[i][j] * vv[j][tid];
    o[((size_t)i * T_ + t) * 512 + hh * 64 + tid] = (_Float16)p;
  }
}

// agents_m[b,v,d] = mean over l
__global__ void agents_mean(const float* __restrict__ agents, float* __restrict__ am)
{
  const int i = blockIdx.x * 256 + threadIdx.x;   // B*V*512 = 8192
  const int d = i & 511;
  const int bv = i >> 9;
  const int v = bv & 7, b = bv >> 3;
  float s = 0.f;
  for (int l = 0; l < L_; ++l)
    s += agents[((size_t)v * T_ + (size_t)b * L_ + l) * 512 + d];
  am[i] = s * (1.f / (float)L_);
}

__global__ void featpre_k(const float* __restrict__ am, float* __restrict__ fp)
{
  const int i = blockIdx.x * 256 + threadIdx.x;   // B*512
  const int d = i & 511, b = i >> 9;
  float s = 0.f;
  for (int v = 0; v < 8; ++v) s += am[((size_t)b * 8 + v) * 512 + d];
  fp[i] = s * 0.125f;
}

// per (b,v): aproj = relu(am@w1^T+b1)@w2^T+b2 ; agents_logits = am@acw^T+acb
__global__ __launch_bounds__(256)
void head_bv(const float* __restrict__ am, const float* __restrict__ w1,
             const float* __restrict__ b1, const float* __restrict__ w2,
             const float* __restrict__ b2, const float* __restrict__ acw,
             const float* __restrict__ acb, float* __restrict__ out)
{
  __shared__ float a[512];
  __shared__ float hid[512];
  const int bv = blockIdx.x, tid = threadIdx.x;
  for (int c = tid; c < 512; c += 256) a[c] = am[(size_t)bv * 512 + c];
  __syncthreads();
  for (int hr = tid; hr < 512; hr += 256) {
    float p = b1[hr];
    for (int d = 0; d < 512; ++d) p += a[d] * w1[hr * 512 + d];
    hid[hr] = p > 0.f ? p : 0.f;
  }
  __syncthreads();
  if (tid < 64) {
    float p = b2[tid];
    for (int h2 = 0; h2 < 512; ++h2) p += hid[h2] * w2[tid * 512 + h2];
    out[1032 + bv * 64 + tid] = p;                     // aproj
  } else if (tid < 68) {
    const int oo = tid - 64;
    float p = acb[oo];
    for (int d = 0; d < 512; ++d) p += a[d] * acw[oo * 512 + d];
    out[2056 + bv * 4 + oo] = p;                       // agents_logits
  }
}

// feat = LN(featpre, 1e-6); logits; softmax
__global__ __launch_bounds__(256)
void feat_k(const float* __restrict__ fp, const float* __restrict__ g,
            const float* __restrict__ b, const float* __restrict__ cw,
            const float* __restrict__ cb, float* __restrict__ out)
{
  __shared__ float row[512];
  __shared__ float feat[512];
  __shared__ float red[8];
  __shared__ float lg[4];
  const int bb = blockIdx.x, tid = threadIdx.x;
  float part = 0.f;
  for (int c = tid; c < 512; c += 256) { const float v = fp[bb * 512 + c]; row[c] = v; part += v; }
  const float mu = block_sum256(part, red) * (1.f / 512.f);
  float vp = 0.f;
  for (int c = tid; c < 512; c += 256) { const float d0 = row[c] - mu; vp += d0 * d0; }
  const float var = block_sum256(vp, red) * (1.f / 512.f);
  const float inv = rsqrtf(var + 1e-6f);
  for (int c = tid; c < 512; c += 256) {
    const float f = (row[c] - mu) * inv * g[c] + b[c];
    feat[c] = f;
    out[bb * 512 + c] = f;
  }
  __syncthreads();
  if (tid < 4) {
    float p = cb[tid];
    for (int d = 0; d < 512; ++d) p += feat[d] * cw[tid * 512 + d];
    lg[tid] = p;
    out[1024 + bb * 4 + tid] = p;
  }
  __syncthreads();
  if (tid == 0) {
    const float m = fmaxf(fmaxf(lg[0], lg[1]), fmaxf(lg[2], lg[3]));
    const float e0 = __expf(lg[0] - m), e1 = __expf(lg[1] - m);
    const float e2 = __expf(lg[2] - m), e3 = __expf(lg[3] - m);
    const float s = 1.f / (e0 + e1 + e2 + e3);
    out[2120 + bb * 4 + 0] = e0 * s;
    out[2120 + bb * 4 + 1] = e1 * s;
    out[2120 + bb * 4 + 2] = e2 * s;
    out[2120 + bb * 4 + 3] = e3 * s;
  }
}

// ---------------------------------------------------------------------------
extern "C" void kernel_launch(void* const* d_in, const int* in_sizes, int n_in,
                              void* d_out, int out_size, void* d_ws, size_t ws_size,
                              hipStream_t stream)
{
  (void)in_sizes; (void)n_in; (void)out_size; (void)ws_size;

  const float* x     = (const float*)d_in[0];
  const float* bn_g  = (const float*)d_in[1];
  const float* bn_b  = (const float*)d_in[2];
  const float* emb_w = (const float*)d_in[3];
  const float* emb_b = (const float*)d_in[4];
  const float* eln_g = (const float*)d_in[5];
  const float* eln_b = (const float*)d_in[6];
  // layer0 mamba = 7..17, layer0 attn = 18..21, layer1 mamba = 22..32
  const float* fln_g = (const float*)d_in[33];
  const float* fln_b = (const float*)d_in[34];
  const float* cls_w = (const float*)d_in[35];
  const float* cls_b = (const float*)d_in[36];
  const float* acw   = (const float*)d_in[37];
  const float* acb   = (const float*)d_in[38];
  const float* pw1   = (const float*)d_in[39];
  const float* pb1   = (const float*)d_in[40];
  const float* pw2   = (const float*)d_in[41];
  const float* pb2   = (const float*)d_in[42];

  // workspace allocator (256B aligned)
  char* wsp = (char*)d_ws;
  size_t cur = 0;
  auto alloc = [&](size_t bytes) -> void* {
    void* p = wsp + cur;
    cur += (bytes + 255) & ~(size_t)255;
    return p;
  };

  float*    xz     = (float*)   alloc(sizeof(float) * (size_t)V_ * T_ * 2048); // aliases: xe, qkv
  float*    h      = (float*)   alloc(sizeof(float) * (size_t)V_ * T_ * 512);
  _Float16* hf16   = (_Float16*)alloc(2 * (size_t)V_ * T_ * 512);              // alias: agents_f16
  _Float16* xf16   = (_Float16*)alloc(2 * (size_t)R0_ * 1024);
  float*    xc     = (float*)   alloc(sizeof(float) * (size_t)V_ * T_ * 1024);
  float*    dbc    = (float*)   alloc(sizeof(float) * (size_t)V_ * T_ * 40);
  float*    dlt    = (float*)   alloc(sizeof(float) * (size_t)V_ * T_ * 1024);
  _Float16* yf16   = (_Float16*)alloc(2 * (size_t)V_ * T_ * 1024);             // alias: attn O f16
  float*    agents = (float*)   alloc(sizeof(float) * (size_t)V_ * T_ * 512);
  float*    am     = (float*)   alloc(sizeof(float) * (size_t)B_ * V_ * 512);
  float*    fp     = (float*)   alloc(sizeof(float) * (size_t)B_ * 512);
  _Float16* w_emb  = (_Float16*)alloc(2 * (size_t)1024 * 1024);
  _Float16* w_in0  = (_Float16*)alloc(2 * (size_t)8 * 2048 * 512);
  _Float16* w_in1  = (_Float16*)alloc(2 * (size_t)8 * 2048 * 512);
  _Float16* w_out0 = (_Float16*)alloc(2 * (size_t)8 * 512 * 1024);
  _Float16* w_out1 = (_Float16*)alloc(2 * (size_t)8 * 512 * 1024);
  _Float16* w_qkv  = (_Float16*)alloc(2 * (size_t)1536 * 512);
  _Float16* w_ao   = (_Float16*)alloc(2 * (size_t)512 * 512);

  auto cvt = [&](const float* s, _Float16* d, size_t n) {
    f32_to_f16<<<dim3((unsigned)((n + 255) / 256)), dim3(256), 0, stream>>>(s, d, (int)n);
  };
  auto gemm = [&](const _Float16* A, const _Float16* W, float* C, const float* bias,
                  int M, int N, int K, int batch, long long sA, long long sW, long long sC) {
    dim3 grid((unsigned)(N / 128), (unsigned)(M / 64), (unsigned)batch);
    gemm_f16f32<<<grid, dim3(256), 0, stream>>>(A, W, C, bias, M, N, K, sA, sW, sC);
  };

  // weight conversions (deterministic each call)
  cvt(emb_w, w_emb, (size_t)1024 * 1024);
  cvt((const float*)d_in[7 + 2],  w_in0,  (size_t)8 * 2048 * 512);
  cvt((const float*)d_in[7 + 10], w_out0, (size_t)8 * 512 * 1024);
  cvt((const float*)d_in[22 + 2], w_in1,  (size_t)8 * 2048 * 512);
  cvt((const float*)d_in[22 + 10], w_out1, (size_t)8 * 512 * 1024);
  cvt((const float*)d_in[18], w_qkv, (size_t)1536 * 512);
  cvt((const float*)d_in[20], w_ao,  (size_t)512 * 512);

  // embedding: bn -> f16 -> GEMM -> (+bias, LN, GLU) -> h (V,T,512)
  prep_x<<<dim3((unsigned)((size_t)R0_ * 1024 / 256)), dim3(256), 0, stream>>>(x, bn_g, bn_b, xf16);
  gemm(xf16, w_emb, xz, nullptr, R0_, 1024, 1024, 1, 0, 0, 0);
  embed_post<<<dim3(R0_), dim3(256), 0, stream>>>(xz, emb_b, eln_g, eln_b, h);

  for (int li = 0; li < 2; ++li) {
    const int mb = (li == 0) ? 7 : 22;
    const float* ln_g    = (const float*)d_in[mb + 0];
    const float* ln_b    = (const float*)d_in[mb + 1];
    const float* conv_w  = (const float*)d_in[mb + 3];
    const float* conv_b  = (const float*)d_in[mb + 4];
    const float* xproj_w = (const float*)d_in[mb + 5];
    const float* dt_w    = (const float*)d_in[mb + 6];
    const float* dt_b    = (const float*)d_in[mb + 7];
    const float* A_log   = (const float*)d_in[mb + 8];
    const float* Dp      = (const float*)d_in[mb + 9];
    _Float16* w_in  = (li == 0) ? w_in0  : w_in1;
    _Float16* w_out = (li == 0) ? w_out0 : w_out1;

    ln512_f16<<<dim3(V_ * T_), dim3(256), 0, stream>>>(h, ln_g, ln_b, hf16);
    gemm(hf16, w_in, xz, nullptr, T_, 2048, 512, V_,
         (long long)T_ * 512, (long long)2048 * 512, (long long)T_ * 2048);
    conv_silu<<<dim3((unsigned)((size_t)V_ * T_ * 1024 / 256)), dim3(256), 0, stream>>>(
        xz, conv_w, conv_b, xc);
    xproj_k<<<dim3(V_ * T_), dim3(256), 0, stream>>>(xc, xproj_w, dbc);
    delta_k<<<dim3((unsigned)((size_t)V_ * T_ * 1024 / 256)), dim3(256), 0, stream>>>(
        dbc, dt_w, dt_b, dlt);
    scan_k<<<dim3(V_ * B_ * 1024 / 256), dim3(256), 0, stream>>>(
        dlt, xc, dbc, xz, A_log, Dp, yf16);
    gemm(yf16, w_out, agents, nullptr, T_, 512, 1024, V_,
         (long long)T_ * 1024, (long long)512 * 1024, (long long)T_ * 512);

    if (li == 0) {
      const float* in_b  = (const float*)d_in[19];
      const float* out_b = (const float*)d_in[21];
      f32_to_f16<<<dim3((unsigned)((size_t)V_ * T_ * 512 / 256)), dim3(256), 0, stream>>>(
          agents, hf16, (int)((size_t)V_ * T_ * 512));
      gemm(hf16, w_qkv, xz, in_b, V_ * T_, 1536, 512, 1, 0, 0, 0);       // qkv (aliases xz)
      attn_k<<<dim3(T_, 8), dim3(64), 0, stream>>>(xz, yf16);            // O f16 (aliases yf16)
      gemm(yf16, w_ao, h, out_b, V_ * T_, 512, 512, 1, 0, 0, 0);         // h for layer 1
    }
  }

  // heads
  float* out = (float*)d_out;
  agents_mean<<<dim3(B_ * V_ * 512 / 256), dim3(256), 0, stream>>>(agents, am);
  featpre_k<<<dim3(B_ * 512 / 256), dim3(256), 0, stream>>>(am, fp);
  feat_k<<<dim3(B_), dim3(256), 0, stream>>>(fp, fln_g, fln_b, cls_w, cls_b, out);
  head_bv<<<dim3(B_ * V_), dim3(256), 0, stream>>>(am, pw1, pb1, pw2, pb2, acw, acb, out);
}
